// SoftDTW_34265249087925
// MI455X (gfx1250) — compile-verified, matrix operand order checked by reading, not run on hardware
//
#include <hip/hip_runtime.h>
#include <hip/hip_bf16.h>

typedef __attribute__((ext_vector_type(2)))  float     v2f;
typedef __attribute__((ext_vector_type(8)))  float     v8f;
typedef __attribute__((ext_vector_type(16))) _Float16  v16h;

constexpr int   kB = 16, kN = 1024, kM = 1024, kD = 128;
constexpr float kBIG = 1e10f;

// ---------------------------------------------------------------------------
// Kernel 1: row-normalize x and y.  One wave32 per 128-float vector:
// each lane loads a float4 (32*4 = 128), wave-reduces sum of squares,
// scales, stores.  Matches reference: v / (||v|| + 1e-8).
// ---------------------------------------------------------------------------
__global__ void __launch_bounds__(256)
normalize_kernel(const float* __restrict__ x, const float* __restrict__ y,
                 float* __restrict__ xn, float* __restrict__ yn, int nx_vec)
{
    const int wave = blockIdx.x * (blockDim.x >> 5) + (threadIdx.x >> 5);
    const int lane = threadIdx.x & 31;

    const float* src;
    float*       dst;
    if (wave < nx_vec) {
        src = x  + (size_t)wave * kD;
        dst = xn + (size_t)wave * kD;
    } else {
        const int w = wave - nx_vec;
        src = y  + (size_t)w * kD;
        dst = yn + (size_t)w * kD;
    }

    float4 v = ((const float4*)src)[lane];
    float ss = v.x * v.x + v.y * v.y + v.z * v.z + v.w * v.w;
#pragma unroll
    for (int m = 16; m >= 1; m >>= 1)
        ss += __shfl_xor(ss, m, 32);

    const float inv = 1.0f / (sqrtf(ss) + 1e-8f);
    v.x *= inv; v.y *= inv; v.z *= inv; v.w *= inv;
    ((float4*)dst)[lane] = v;
}

// ---------------------------------------------------------------------------
// Kernel 2: D[b] = 1 - Xn[b] (1024x128) * Yn[b]^T (128x1024) via WMMA.
// One wave computes a 16(M) x 64(N) strip: 4 f32 accumulators, A reused 4x.
//
// fp32 A 16x4 fragment (ISA 7.12.2): lanes 0-15 hold K={k0,k0+1},
// lanes 16-31 hold K={k0+2,k0+3}  -> contiguous float2 per lane.
// B 4x16 fragment mirrors this with N = lane&15 (row-major Yn => float2).
// C/D: v8f element r -> (M = r + 8*(lane>>4), N = lane&15).
// ---------------------------------------------------------------------------
#if __has_builtin(__builtin_amdgcn_wmma_f32_16x16x4_f32)
#define USE_F32_WMMA 1
#else
#define USE_F32_WMMA 0
__device__ inline v16h load_frag_f16(const float* __restrict__ row, int k0, int hi)
{
    // 16-bit A/B 16x32 fragment: lanes 0-15: K = k0+[0..7], k0+16+[0..7]
    //                            lanes16-31: K = k0+8+[0..7], k0+24+[0..7]
    v16h f;
    const float* p0 = row + k0 + hi * 8;
    const float* p1 = row + k0 + 16 + hi * 8;
#pragma unroll
    for (int e = 0; e < 8; ++e) {
        f[e]     = (_Float16)p0[e];
        f[8 + e] = (_Float16)p1[e];
    }
    return f;
}
#endif

__global__ void __launch_bounds__(256)
cosdist_wmma_kernel(const float* __restrict__ Xn, const float* __restrict__ Yn,
                    float* __restrict__ D)
{
    const int wave = blockIdx.x * (blockDim.x >> 5) + (threadIdx.x >> 5);
    const int lane = threadIdx.x & 31;
    const int lo   = lane & 15;
    const int hi   = lane >> 4;

    constexpr int tiles_n = kM / 64;                 // 16 strips of width 64
    constexpr int tiles_per_batch = (kN / 16) * tiles_n;  // 1024
    const int b  = wave / tiles_per_batch;
    const int rw = wave % tiles_per_batch;
    const int m0 = (rw / tiles_n) * 16;
    const int n0 = (rw % tiles_n) * 64;

    const float* X = Xn + ((size_t)b * kN + m0) * kD;
    const float* Y = Yn + ((size_t)b * kM + n0) * kD;

    v8f acc0 = {}, acc1 = {}, acc2 = {}, acc3 = {};

#if USE_F32_WMMA
    const float* ap  = X + (size_t)lo * kD + 2 * hi;        // A row m0+lo
    const float* bp0 = Y + (size_t)(lo)      * kD + 2 * hi; // B cols n0+lo+{0,16,32,48}
    const float* bp1 = Y + (size_t)(lo + 16) * kD + 2 * hi;
    const float* bp2 = Y + (size_t)(lo + 32) * kD + 2 * hi;
    const float* bp3 = Y + (size_t)(lo + 48) * kD + 2 * hi;

#pragma unroll 4
    for (int k0 = 0; k0 < kD; k0 += 4) {
        const v2f a  = *(const v2f*)(ap  + k0);
        const v2f b0 = *(const v2f*)(bp0 + k0);
        const v2f b1 = *(const v2f*)(bp1 + k0);
        const v2f b2 = *(const v2f*)(bp2 + k0);
        const v2f b3 = *(const v2f*)(bp3 + k0);
        acc0 = __builtin_amdgcn_wmma_f32_16x16x4_f32(false, a, false, b0, (short)0, acc0, false, false);
        acc1 = __builtin_amdgcn_wmma_f32_16x16x4_f32(false, a, false, b1, (short)0, acc1, false, false);
        acc2 = __builtin_amdgcn_wmma_f32_16x16x4_f32(false, a, false, b2, (short)0, acc2, false, false);
        acc3 = __builtin_amdgcn_wmma_f32_16x16x4_f32(false, a, false, b3, (short)0, acc3, false, false);
    }
#else
    const float* arow  = X + (size_t)lo * kD;
    const float* brow0 = Y + (size_t)(lo)      * kD;
    const float* brow1 = Y + (size_t)(lo + 16) * kD;
    const float* brow2 = Y + (size_t)(lo + 32) * kD;
    const float* brow3 = Y + (size_t)(lo + 48) * kD;
    for (int k0 = 0; k0 < kD; k0 += 32) {
        const v16h a  = load_frag_f16(arow,  k0, hi);
        const v16h b0 = load_frag_f16(brow0, k0, hi);
        const v16h b1 = load_frag_f16(brow1, k0, hi);
        const v16h b2 = load_frag_f16(brow2, k0, hi);
        const v16h b3 = load_frag_f16(brow3, k0, hi);
        acc0 = __builtin_amdgcn_wmma_f32_16x16x32_f16(false, a, false, b0, (short)0, acc0, false, false);
        acc1 = __builtin_amdgcn_wmma_f32_16x16x32_f16(false, a, false, b1, (short)0, acc1, false, false);
        acc2 = __builtin_amdgcn_wmma_f32_16x16x32_f16(false, a, false, b2, (short)0, acc2, false, false);
        acc3 = __builtin_amdgcn_wmma_f32_16x16x32_f16(false, a, false, b3, (short)0, acc3, false, false);
    }
#endif

    float* Dt = D + (size_t)b * kN * kM;
#pragma unroll
    for (int r = 0; r < 8; ++r) {
        const int    m   = m0 + r + 8 * hi;
        const size_t row = (size_t)m * kM + n0 + lo;
        Dt[row]      = 1.0f - acc0[r];
        Dt[row + 16] = 1.0f - acc1[r];
        Dt[row + 32] = 1.0f - acc2[r];
        Dt[row + 48] = 1.0f - acc3[r];
    }
}

// ---------------------------------------------------------------------------
// Kernel 3: soft-DTW DP, one 1024-thread workgroup per batch.
// Anti-diagonal wavefront, three rotating diagonals in LDS, one barrier per
// diagonal (2047 total).  Thread t computes row i = t+1.
// R[i][j] = D[i-1][j-1] + softmin(R[i-1][j-1], R[i-1][j], R[i][j-1])
// softmin(a,b,c) = mn - log(e^{mn-a} + e^{mn-b} + e^{mn-c}),  gamma = 1.
// ---------------------------------------------------------------------------
__global__ void __launch_bounds__(1024)
softdtw_kernel(const float* __restrict__ D, float* __restrict__ out)
{
    __shared__ float buf[3][kN + 2];

    const int b = blockIdx.x;
    const int t = threadIdx.x;     // 0..1023
    const int i = t + 1;           // DP row 1..1024

    const float* Drow = D + (size_t)b * kN * kM + (size_t)(i - 1) * kM;

    // diag s=0: R[0][0]=0, rest invalid(BIG).  diag s=1: all BIG.
    buf[0][t] = (t == 0) ? 0.0f : kBIG;
    buf[1][t] = kBIG;
    if (t == 0) {
        buf[0][kN] = kBIG;
        buf[1][kN] = kBIG;
    }
    __syncthreads();

    float r = kBIG;
    for (int s = 2; s <= kN + kM; ++s) {
        float*       cur = buf[s % 3];
        const float* p1  = buf[(s + 2) % 3];   // diag s-1
        const float* p2  = buf[(s + 1) % 3];   // diag s-2

        const int j = s - i;
        float val = kBIG;
        if (j >= 1 && j <= kM) {
            const float a  = p2[i - 1];        // R[i-1][j-1]
            const float bu = p1[i - 1];        // R[i-1][j]
            const float cl = p1[i];            // R[i][j-1]
            const float mn = fminf(a, fminf(bu, cl));
            const float sum = __expf(mn - a) + __expf(mn - bu) + __expf(mn - cl);
            const float sm  = mn - __logf(sum);
            val = Drow[j - 1] + sm;
            r = val;
        }
        cur[i] = val;
        if (t == 0) cur[0] = kBIG;             // boundary R[0][j] = BIG
        __syncthreads();
    }

    if (t == kN - 1) out[b] = r;               // R[N][M]
}

// ---------------------------------------------------------------------------
extern "C" void kernel_launch(void* const* d_in, const int* in_sizes, int n_in,
                              void* d_out, int out_size, void* d_ws, size_t ws_size,
                              hipStream_t stream)
{
    const float* x = (const float*)d_in[0];   // [16,1024,128] f32
    const float* y = (const float*)d_in[1];   // [16,1024,128] f32
    float* out = (float*)d_out;               // [16] f32

    char* ws = (char*)d_ws;
    constexpr size_t xn_bytes = (size_t)kB * kN * kD * sizeof(float);  // 8 MB
    constexpr size_t yn_bytes = (size_t)kB * kM * kD * sizeof(float);  // 8 MB
    float* xn = (float*)ws;
    float* yn = (float*)(ws + xn_bytes);
    float* D  = (float*)(ws + xn_bytes + yn_bytes);                    // 64 MB

    // 1) normalize: 32768 vectors, 8 waves/block -> 4096 blocks
    {
        const int nvec   = kB * kN + kB * kM;
        const int blocks = nvec / 8;
        normalize_kernel<<<blocks, 256, 0, stream>>>(x, y, xn, yn, kB * kN);
    }

    // 2) distance matrix via WMMA: 16384 waves (16x64 strip each), 8/block
    {
        const int waves  = kB * (kN / 16) * (kM / 64);
        const int blocks = waves / 8;
        cosdist_wmma_kernel<<<blocks, 256, 0, stream>>>(xn, yn, D);
    }

    // 3) soft-DTW DP: one workgroup (32 wave32s) per batch
    softdtw_kernel<<<kB, 1024, 0, stream>>>(D, out);
}
